// AGNN_62689342652830
// MI455X (gfx1250) — matrix operand order repaired and checked by reading, not compiled, not run on hardware
//
#include <hip/hip_runtime.h>

typedef __attribute__((ext_vector_type(16))) _Float16 v16h;
typedef __attribute__((ext_vector_type(8)))  float    v8f;

// ---- float <-> order-preserving uint (for atomic segment-max over floats) ----
__device__ __forceinline__ unsigned f2ord(float f) {
  unsigned u = __float_as_uint(f);
  return (u & 0x80000000u) ? ~u : (u | 0x80000000u);
}
__device__ __forceinline__ float ord2f(unsigned u) {
  return __uint_as_float((u & 0x80000000u) ? (u & 0x7FFFFFFFu) : ~u);
}

__device__ __forceinline__ _Float16 cvt_h(float x, int relu) {
  return (_Float16)(relu ? fmaxf(x, 0.f) : x);
}

// ============================================================================
// GEMM: H[m][n] = sum_k act(A[m][k]) * W[n][k] + bias[n],   K fixed at 128.
// A: [M x 128] f32 row-major, W: [ncols x 128] f32 row-major, H: [M x ncols].
// One wave32 per 16x16 tile; fully unrolled K -> 4x v_wmma_f32_16x16x32_f16.
// Per-lane fragment loads are contiguous -> global_load_b128:
//   A frag = floats [kt+16g .. +15] of row (lane&15)
//   B frag = floats [kt+16g .. +15] of W row (lane&15)   (B[k][n] = W[n][k])
// ============================================================================
__global__ __launch_bounds__(128) void gemm_wmma_kernel(
    const float* __restrict__ A, const float* __restrict__ W,
    const float* __restrict__ bias, float* __restrict__ H,
    int ncols, int relu_in)
{
  constexpr int K = 128;
  const int lane = threadIdx.x & 31;
  const int wv   = threadIdx.x >> 5;
  const int mt   = blockIdx.x;
  const int nt   = blockIdx.y * 4 + wv;
  const int g    = lane >> 4;     // lane-half select
  const int r16  = lane & 15;

  const float* Arow = A + (size_t)(mt * 16 + r16) * K;
  const float* Wrow = W + (size_t)(nt * 16 + r16) * K;

  v8f acc = {};
#pragma unroll
  for (int kt = 0; kt < K; kt += 32) {
    const float4* ap = (const float4*)(Arow + kt + 16 * g);
    const float4  al0 = ap[0], al1 = ap[1];     // halves a[0..7]
    const float4  ah0 = ap[2], ah1 = ap[3];     // halves a[8..15] (+8 floats)
    const float4* bp = (const float4*)(Wrow + kt + 16 * g);
    const float4  b0 = bp[0], b1 = bp[1], b2 = bp[2], b3 = bp[3];

    v16h a, b;
    a[0]  = cvt_h(al0.x, relu_in); a[1]  = cvt_h(al0.y, relu_in);
    a[2]  = cvt_h(al0.z, relu_in); a[3]  = cvt_h(al0.w, relu_in);
    a[4]  = cvt_h(al1.x, relu_in); a[5]  = cvt_h(al1.y, relu_in);
    a[6]  = cvt_h(al1.z, relu_in); a[7]  = cvt_h(al1.w, relu_in);
    a[8]  = cvt_h(ah0.x, relu_in); a[9]  = cvt_h(ah0.y, relu_in);
    a[10] = cvt_h(ah0.z, relu_in); a[11] = cvt_h(ah0.w, relu_in);
    a[12] = cvt_h(ah1.x, relu_in); a[13] = cvt_h(ah1.y, relu_in);
    a[14] = cvt_h(ah1.z, relu_in); a[15] = cvt_h(ah1.w, relu_in);

    b[0]  = (_Float16)b0.x; b[1]  = (_Float16)b0.y;
    b[2]  = (_Float16)b0.z; b[3]  = (_Float16)b0.w;
    b[4]  = (_Float16)b1.x; b[5]  = (_Float16)b1.y;
    b[6]  = (_Float16)b1.z; b[7]  = (_Float16)b1.w;
    b[8]  = (_Float16)b2.x; b[9]  = (_Float16)b2.y;
    b[10] = (_Float16)b2.z; b[11] = (_Float16)b2.w;
    b[12] = (_Float16)b3.x; b[13] = (_Float16)b3.y;
    b[14] = (_Float16)b3.z; b[15] = (_Float16)b3.w;

    acc = __builtin_amdgcn_wmma_f32_16x16x32_f16(
        /*neg_a=*/false, a, /*neg_b=*/false, b,
        /*c_mod=*/(short)0, acc, /*reuse_a=*/false, /*reuse_b=*/false);
  }
  // D layout: VGPR r holds M = 8g + r, N = lane&15
  const int col = nt * 16 + r16;
  const float bv = bias[col];
#pragma unroll
  for (int r = 0; r < 8; ++r) {
    int m = mt * 16 + 8 * g + r;
    H[(size_t)m * ncols + col] = acc[r] + bv;
  }
}

// ============================================================================
// Per-node inverse L2 norm: rn[i] = 1 / max(||h_i||, 1e-12). One wave per node.
// ============================================================================
__global__ void rownorm_kernel(const float* __restrict__ H, float* __restrict__ rn,
                               int n, int D)
{
  int lane = threadIdx.x & 31;
  int node = (int)(((long long)blockIdx.x * blockDim.x + threadIdx.x) >> 5);
  if (node >= n) return;
  const float* hp = H + (size_t)node * D;
  float p;
  if (D == 128) {
    float4 v = ((const float4*)hp)[lane];
    p = v.x * v.x + v.y * v.y + v.z * v.z + v.w * v.w;
  } else {
    float2 v = ((const float2*)hp)[lane];
    p = v.x * v.x + v.y * v.y;
  }
#pragma unroll
  for (int o = 16; o > 0; o >>= 1) p += __shfl_xor(p, o, 32);
  if (lane == 0) rn[node] = 1.f / fmaxf(sqrtf(p), 1e-12f);
}

// ============================================================================
// Zero segment state + aggregation target.
// ============================================================================
__global__ void init_kernel(unsigned* __restrict__ segmax, float* __restrict__ segsum,
                            float* __restrict__ agg, int n, int D)
{
  long long i = (long long)blockIdx.x * blockDim.x + threadIdx.x;
  if (i < (long long)n * D) agg[i] = 0.f;
  if (i < n) { segmax[i] = 0u; segsum[i] = 0.f; }   // ord-uint 0 == "below all floats"
}

// ============================================================================
// Per-edge cosine similarity + atomic segment max. One wave per edge.
// Edges e in [0,E) from arrays; e in [E,ET) are self-loops (s=d=e-E).
// ============================================================================
__global__ void edge_alpha_kernel(const float* __restrict__ H, const float* __restrict__ rn,
    const int* __restrict__ src, const int* __restrict__ dst,
    float* __restrict__ alpha, unsigned* __restrict__ segmax,
    int E, int ET, int D)
{
  int lane = threadIdx.x & 31;
  int e = (int)(((long long)blockIdx.x * blockDim.x + threadIdx.x) >> 5);
  if (e >= ET) return;
  int s, d;
  if (e < E) { s = src[e]; d = dst[e]; } else { s = d = e - E; }
  const float* hs = H + (size_t)s * D;
  const float* hd = H + (size_t)d * D;
  float p;
  if (D == 128) {
    float4 a = ((const float4*)hs)[lane];
    float4 b = ((const float4*)hd)[lane];
    p = a.x * b.x + a.y * b.y + a.z * b.z + a.w * b.w;
  } else {
    float2 a = ((const float2*)hs)[lane];
    float2 b = ((const float2*)hd)[lane];
    p = a.x * b.x + a.y * b.y;
  }
#pragma unroll
  for (int o = 16; o > 0; o >>= 1) p += __shfl_xor(p, o, 32);
  if (lane == 0) {
    float al = p * rn[s] * rn[d];
    alpha[e] = al;
    atomicMax(&segmax[d], f2ord(al));
  }
}

// ============================================================================
// ex = exp(alpha - segmax[dst]) (in place), segsum[dst] += ex. Thread per edge.
// ============================================================================
__global__ void edge_exp_kernel(float* __restrict__ alpha,
    const unsigned* __restrict__ segmax, float* __restrict__ segsum,
    const int* __restrict__ dst, int E, int ET)
{
  int e = (int)((long long)blockIdx.x * blockDim.x + threadIdx.x);
  if (e >= ET) return;
  int d = (e < E) ? dst[e] : (e - E);
  float ex = __expf(alpha[e] - ord2f(segmax[d]));
  alpha[e] = ex;
  atomicAdd(&segsum[d], ex);
}

// ============================================================================
// agg[dst] += h[src] * (ex / segsum[dst]). One wave per edge, float atomics.
// ============================================================================
__global__ void edge_scatter_kernel(const float* __restrict__ H,
    const float* __restrict__ alpha, const float* __restrict__ segsum,
    const int* __restrict__ src, const int* __restrict__ dst,
    float* __restrict__ agg, int E, int ET, int D)
{
  int lane = threadIdx.x & 31;
  int e = (int)(((long long)blockIdx.x * blockDim.x + threadIdx.x) >> 5);
  if (e >= ET) return;
  int s, d;
  if (e < E) { s = src[e]; d = dst[e]; } else { s = d = e - E; }
  float a = alpha[e] / segsum[d];
  const float* hs = H + (size_t)s * D;
  float* op = agg + (size_t)d * D;
  if (D == 128) {
    float4 v = ((const float4*)hs)[lane];
    atomicAdd(op + lane * 4 + 0, v.x * a);
    atomicAdd(op + lane * 4 + 1, v.y * a);
    atomicAdd(op + lane * 4 + 2, v.z * a);
    atomicAdd(op + lane * 4 + 3, v.w * a);
  } else {
    float2 v = ((const float2*)hs)[lane];
    atomicAdd(op + lane * 2 + 0, v.x * a);
    atomicAdd(op + lane * 2 + 1, v.y * a);
  }
}

// ============================================================================
// log_softmax over 64 classes. One wave per node (2 values per lane).
// ============================================================================
__global__ void logsoftmax_kernel(const float* __restrict__ Hin,
                                  float* __restrict__ out, int n)
{
  int lane = threadIdx.x & 31;
  int node = (int)(((long long)blockIdx.x * blockDim.x + threadIdx.x) >> 5);
  if (node >= n) return;
  float2 v = ((const float2*)(Hin + (size_t)node * 64))[lane];
  float m = fmaxf(v.x, v.y);
#pragma unroll
  for (int o = 16; o > 0; o >>= 1) m = fmaxf(m, __shfl_xor(m, o, 32));
  float s = expf(v.x - m) + expf(v.y - m);
#pragma unroll
  for (int o = 16; o > 0; o >>= 1) s += __shfl_xor(s, o, 32);
  float ls = logf(s);
  float2 r; r.x = v.x - m - ls; r.y = v.y - m - ls;
  ((float2*)(out + (size_t)node * 64))[lane] = r;
}

// ============================================================================
extern "C" void kernel_launch(void* const* d_in, const int* in_sizes, int n_in,
                              void* d_out, int out_size, void* d_ws, size_t ws_size,
                              hipStream_t stream)
{
  (void)n_in; (void)out_size; (void)ws_size;
  const float* x  = (const float*)d_in[0];
  const int*   ei = (const int*)d_in[1];
  const float* W1 = (const float*)d_in[2];
  const float* b1 = (const float*)d_in[3];
  const float* W2 = (const float*)d_in[4];
  const float* b2 = (const float*)d_in[5];
  const float* W3 = (const float*)d_in[6];
  const float* b3 = (const float*)d_in[7];

  const int D  = 128;
  const int N  = in_sizes[0] / D;
  const int E  = in_sizes[1] / 2;
  const int ET = E + N;                      // + self loops
  const int* src = ei;
  const int* dst = ei + E;

  // workspace layout (floats): buf0 | buf1 | rn | alpha | segmax | segsum
  float*    buf0   = (float*)d_ws;                  // N*128 (h)
  float*    buf1   = buf0 + (size_t)N * D;          // N*128 (agg / next input)
  float*    rn     = buf1 + (size_t)N * D;          // N
  float*    alpha  = rn + N;                        // ET
  unsigned* segmax = (unsigned*)(alpha + ET);       // N
  float*    segsum = (float*)(segmax + N);          // N

  auto cdiv = [](long long a, long long b) { return (int)((a + b - 1) / b); };

  auto layer = [&](const float* in, const float* W, const float* b, int Dout,
                   int relu_in, float* h, float* out) {
    dim3 gg(N / 16, Dout / 64);
    gemm_wmma_kernel<<<gg, 128, 0, stream>>>(in, W, b, h, Dout, relu_in);
    rownorm_kernel<<<cdiv((long long)N * 32, 256), 256, 0, stream>>>(h, rn, N, Dout);
    init_kernel<<<cdiv((long long)N * Dout, 256), 256, 0, stream>>>(segmax, segsum, out, N, Dout);
    edge_alpha_kernel<<<cdiv((long long)ET * 32, 256), 256, 0, stream>>>(
        h, rn, src, dst, alpha, segmax, E, ET, Dout);
    edge_exp_kernel<<<cdiv((long long)ET, 256), 256, 0, stream>>>(
        alpha, segmax, segsum, dst, E, ET);
    edge_scatter_kernel<<<cdiv((long long)ET * 32, 256), 256, 0, stream>>>(
        h, alpha, segsum, src, dst, out, E, ET, Dout);
  };

  layer(x,    W1, b1, 128, /*relu_in=*/0, buf0, buf1);
  layer(buf1, W2, b2, 128, /*relu_in=*/1, buf0, buf1);
  layer(buf1, W3, b3,  64, /*relu_in=*/1, buf0, buf1);
  logsoftmax_kernel<<<cdiv((long long)N * 32, 256), 256, 0, stream>>>(buf1, (float*)d_out, N);
}
